// StructureAttention_56255481643637
// MI455X (gfx1250) — matrix operand order repaired and checked by reading, not compiled, not run on hardware
//
#include <hip/hip_runtime.h>
#include <hip/hip_bf16.h>
#include <math.h>

typedef __bf16 bf16;
typedef bf16  v8bf  __attribute__((ext_vector_type(8)));
typedef bf16  v16bf __attribute__((ext_vector_type(16)));
typedef float v8f   __attribute__((ext_vector_type(8)));

#define NB 4
#define LL 8192
#define SSEQ 8192
#define CC 256
#define HH 8
#define DD 32
#define MTOK (NB * LL) // 32768
#define EPS_ATTN 1e-6f
#define EPS_LN 1e-5f

// ---------------------------------------------------------------------------
// Conversions
// ---------------------------------------------------------------------------
__global__ __launch_bounds__(256) void k_convert_wt(const float* __restrict__ W,
                                                    bf16* __restrict__ Wt,
                                                    int K, int N) {
  int i = blockIdx.x * 256 + threadIdx.x;
  if (i < K * N) {
    int n = i / K, k = i % K;              // Wt[n][k] = W[k][n]
    Wt[i] = (bf16)W[(size_t)k * N + n];
  }
}

__global__ __launch_bounds__(256) void k_convert_x_to_hcat(const float* __restrict__ x,
                                                           bf16* __restrict__ hcat) {
  long i = (long)blockIdx.x * 256 + threadIdx.x; // over MTOK*CC
  if (i < (long)MTOK * CC) {
    long r = i >> 8;
    int  c = (int)(i & 255);
    hcat[r * (2 * CC) + c] = (bf16)x[i];
  }
}

__global__ __launch_bounds__(256) void k_convert(const float* __restrict__ s,
                                                 bf16* __restrict__ o, long n) {
  long i = (long)blockIdx.x * 256 + threadIdx.x;
  if (i < n) o[i] = (bf16)s[i];
}

// ---------------------------------------------------------------------------
// WMMA GEMM: C[M x N] = epi( A[M x K](bf16,lda) * Bt[N x K]^T (bf16,ldb) )
// Block tile 128x64, 256 threads (8 waves). Wave (wm 0..3, wn 0..1) computes a
// 32x32 sub-tile with 4 f32 accumulators (4 v_wmma_f32_16x16x32_bf16 / K-step).
// Register-staged double-buffered LDS: 1 barrier per K-step, next tile's
// global loads issued before the WMMAs so VMEM latency overlaps matrix math.
// EPI: 0 = store f32 ; 1 = elu+1 -> bf16 ; 2 = relu -> bf16 ; 3 = *scale -> bf16
// ---------------------------------------------------------------------------
template <int EPI>
__global__ __launch_bounds__(256) void k_gemm(const bf16* __restrict__ A, int lda,
                                              const bf16* __restrict__ Bt, int ldb,
                                              void* __restrict__ Cout, int ldc,
                                              int Kdim, float scale) {
  __shared__ v8bf sA[2][128][5]; // 128 x 32 bf16, row padded to 40 elems
  __shared__ v8bf sB[2][64][5];  // 64 x 32 bf16 (Bt rows)
  const int tid  = threadIdx.x;
  const int lane = tid & 31;
  const int wave = tid >> 5;
  const int wm   = wave >> 1; // 0..3 -> rows wm*32
  const int wn   = wave & 1;  // 0..1 -> cols wn*32

  // cooperative-load indices: A = 2 x 16B per thread, B = 1 x 16B per thread
  const int ar = tid >> 1;        // 0..127
  const int ac = (tid & 1) * 2;   // v8bf col 0 or 2 (K 0-15 / 16-31)
  const int br = tid >> 2;        // 0..63
  const int bc = tid & 3;         // v8bf col 0..3

  const size_t arow = (size_t)((int)blockIdx.x * 128 + ar) * lda;
  const size_t brow = (size_t)((int)blockIdx.y * 64 + br) * ldb;

  v8f acc00 = {}, acc01 = {}, acc10 = {}, acc11 = {};

  const int half = lane >> 4; // ISA K-half select for A/B fragments
  const int loc  = lane & 15;

  // preload K-step 0 into buffer 0
  v8bf ra0 = *(const v8bf*)(A + arow + ac * 8);
  v8bf ra1 = *(const v8bf*)(A + arow + ac * 8 + 8);
  v8bf rb  = *(const v8bf*)(Bt + brow + bc * 8);
  sA[0][ar][ac] = ra0;
  sA[0][ar][ac + 1] = ra1;
  sB[0][br][bc] = rb;
  __syncthreads();

  const int nsteps = Kdim >> 5;
  int buf = 0;
  for (int i = 0; i < nsteps; ++i) {
    if (i + 1 < nsteps) { // issue next tile's global loads before the math
      const int k0 = (i + 1) << 5;
      ra0 = *(const v8bf*)(A + arow + k0 + ac * 8);
      ra1 = *(const v8bf*)(A + arow + k0 + ac * 8 + 8);
      rb  = *(const v8bf*)(Bt + brow + k0 + bc * 8);
    }

    // B fragments (cols wn*32.. and wn*32+16..)
    v16bf b0, b1;
    {
      v8bf t0 = sB[buf][wn * 32 + loc][half];
      v8bf t1 = sB[buf][wn * 32 + loc][2 + half];
      v8bf u0 = sB[buf][wn * 32 + 16 + loc][half];
      v8bf u1 = sB[buf][wn * 32 + 16 + loc][2 + half];
#pragma unroll
      for (int j = 0; j < 8; j++) {
        b0[j] = t0[j]; b0[j + 8] = t1[j];
        b1[j] = u0[j]; b1[j + 8] = u1[j];
      }
    }
    // A fragments (rows wm*32.. and wm*32+16..)
    v16bf a0, a1;
    {
      v8bf t0 = sA[buf][wm * 32 + loc][half];
      v8bf t1 = sA[buf][wm * 32 + loc][2 + half];
      v8bf u0 = sA[buf][wm * 32 + 16 + loc][half];
      v8bf u1 = sA[buf][wm * 32 + 16 + loc][2 + half];
#pragma unroll
      for (int j = 0; j < 8; j++) {
        a0[j] = t0[j]; a0[j + 8] = t1[j];
        a1[j] = u0[j]; a1[j + 8] = u1[j];
      }
    }

    acc00 = __builtin_amdgcn_wmma_f32_16x16x32_bf16(false, a0, false, b0, (short)0, acc00, false, false);
    acc01 = __builtin_amdgcn_wmma_f32_16x16x32_bf16(false, a0, false, b1, (short)0, acc01, false, false);
    acc10 = __builtin_amdgcn_wmma_f32_16x16x32_bf16(false, a1, false, b0, (short)0, acc10, false, false);
    acc11 = __builtin_amdgcn_wmma_f32_16x16x32_bf16(false, a1, false, b1, (short)0, acc11, false, false);

    if (i + 1 < nsteps) { // stage next tile into the other buffer
      sA[buf ^ 1][ar][ac] = ra0;
      sA[buf ^ 1][ar][ac + 1] = ra1;
      sB[buf ^ 1][br][bc] = rb;
    }
    __syncthreads();
    buf ^= 1;
  }

  // Epilogue. C/D layout: lane -> N=lane&15 ; VGPR e -> M=e (lane<16) / 8+e
  const int n0    = (int)blockIdx.y * 64 + wn * 32 + loc;
  const int mbase = (int)blockIdx.x * 128 + wm * 32 + (half ? 8 : 0);
#pragma unroll
  for (int e = 0; e < 8; e++) {
    const int m0 = mbase + e;
    const int m1 = mbase + 16 + e;
    float v00 = acc00[e], v01 = acc01[e], v10 = acc10[e], v11 = acc11[e];
    if (EPI == 0) {
      float* C = (float*)Cout;
      C[(size_t)m0 * ldc + n0]      = v00;
      C[(size_t)m0 * ldc + n0 + 16] = v01;
      C[(size_t)m1 * ldc + n0]      = v10;
      C[(size_t)m1 * ldc + n0 + 16] = v11;
    } else {
      if (EPI == 1) { // elu(x)+1
        v00 = v00 > 0.f ? v00 + 1.f : __expf(v00);
        v01 = v01 > 0.f ? v01 + 1.f : __expf(v01);
        v10 = v10 > 0.f ? v10 + 1.f : __expf(v10);
        v11 = v11 > 0.f ? v11 + 1.f : __expf(v11);
      } else if (EPI == 2) { // relu
        v00 = fmaxf(v00, 0.f); v01 = fmaxf(v01, 0.f);
        v10 = fmaxf(v10, 0.f); v11 = fmaxf(v11, 0.f);
      } else { // scale
        v00 *= scale; v01 *= scale; v10 *= scale; v11 *= scale;
      }
      bf16* C = (bf16*)Cout;
      C[(size_t)m0 * ldc + n0]      = (bf16)v00;
      C[(size_t)m0 * ldc + n0 + 16] = (bf16)v01;
      C[(size_t)m1 * ldc + n0]      = (bf16)v10;
      C[(size_t)m1 * ldc + n0 + 16] = (bf16)v11;
    }
  }
}

// ---------------------------------------------------------------------------
// KV[n,h,d,v] = sum_s K[n,s,h,d] * Vn[n,s,h,v] ; Ksum[n,h,d] = sum_s K
// grid (32 pairs, 32 slices) x 256 threads. Shuffle outer-product + atomics.
// ---------------------------------------------------------------------------
__global__ __launch_bounds__(256) void k_kv(const bf16* __restrict__ Kb,
                                            const bf16* __restrict__ Vb,
                                            float* __restrict__ KV,
                                            float* __restrict__ Ksum) {
  const int pair = blockIdx.x; // n*8 + h
  const int n = pair >> 3, h = pair & 7;
  const int tid = threadIdx.x, lane = tid & 31, wave = tid >> 5;
  const long s0   = (long)blockIdx.y * 256 + wave * 32;
  const long base = ((long)n * SSEQ) * CC + h * 32 + lane;

  float acc[32];
#pragma unroll
  for (int i = 0; i < 32; i++) acc[i] = 0.f;
  float ks = 0.f;

  for (int s = 0; s < 32; s++) {
    const long off = base + (s0 + s) * CC;
    const float kd = (float)Kb[off];
    const float vd = (float)Vb[off];
    ks += kd;
#pragma unroll
    for (int i = 0; i < 32; i++) acc[i] += __shfl(kd, i, 32) * vd;
  }

  __shared__ float red[8][32][32];
  __shared__ float redk[8][32];
#pragma unroll
  for (int i = 0; i < 32; i++) red[wave][i][lane] = acc[i];
  redk[wave][lane] = ks;
  __syncthreads();

  for (int p = tid; p < 1024; p += 256) {
    float s = 0.f;
#pragma unroll
    for (int w = 0; w < 8; w++) s += red[w][p >> 5][p & 31];
    atomicAdd(&KV[(size_t)pair * 1024 + p], s);
  }
  if (tid < 32) {
    float s = 0.f;
#pragma unroll
    for (int w = 0; w < 8; w++) s += redk[w][tid];
    atomicAdd(&Ksum[pair * 32 + tid], s);
  }
}

// ---------------------------------------------------------------------------
// msg[n,l,h,v] = (Q . KV) * S / (Q . Ksum + eps)   -> bf16
// block = (n, 32-token tile); thread = (token, head)
// ---------------------------------------------------------------------------
__global__ __launch_bounds__(256) void k_msg(const bf16* __restrict__ Qb,
                                             const float* __restrict__ KV,
                                             const float* __restrict__ Ksum,
                                             bf16* __restrict__ msgA) {
  const int blk = blockIdx.x;           // n*256 + ltile
  const int n = blk >> 8, ltile = blk & 255;
  const int tid = threadIdx.x;
  const int t  = tid & 31;              // token in tile
  const int hh = tid >> 5;              // head

  __shared__ float sKV[8][32][32];
  __shared__ float sKs[8][32];
  for (int p = tid; p < 8 * 32 * 32; p += 256)
    sKV[p >> 10][(p >> 5) & 31][p & 31] = KV[(size_t)n * 8192 + p];
  for (int p = tid; p < 256; p += 256)
    sKs[p >> 5][p & 31] = Ksum[n * 256 + p];
  __syncthreads();

  const long l  = (long)ltile * 32 + t;
  const bf16* qp = Qb + ((long)n * LL + l) * CC + hh * 32;
  float qv[32];
#pragma unroll
  for (int j = 0; j < 32; j++) qv[j] = (float)qp[j];

  float zden = EPS_ATTN;
#pragma unroll
  for (int j = 0; j < 32; j++) zden += qv[j] * sKs[hh][j];
  const float z = (float)SSEQ / zden;

  bf16* op = msgA + ((long)n * LL + l) * CC + hh * 32;
#pragma unroll
  for (int v = 0; v < 32; v++) {
    float m = 0.f;
#pragma unroll
    for (int j = 0; j < 32; j++) m += qv[j] * sKV[hh][j][v];
    op[v] = (bf16)(m * z);
  }
}

// ---------------------------------------------------------------------------
// LayerNorm over C=256. One wave per row.
// MODE 0: write bf16 into hcat[:, 256:512] (stride 512)
// MODE 1: write f32 out = resid + LN(x)
// ---------------------------------------------------------------------------
template <int MODE>
__global__ __launch_bounds__(256) void k_ln(const float* __restrict__ X,
                                            const float* __restrict__ g,
                                            const float* __restrict__ b,
                                            bf16* __restrict__ outb,
                                            const float* __restrict__ resid,
                                            float* __restrict__ outf) {
  const int tid = threadIdx.x, lane = tid & 31, wave = tid >> 5;
  const long row = (long)blockIdx.x * 8 + wave;
  const float* xp = X + row * CC + lane * 8;
  float v[8];
#pragma unroll
  for (int j = 0; j < 8; j++) v[j] = xp[j];
  float s = 0.f, q = 0.f;
#pragma unroll
  for (int j = 0; j < 8; j++) { s += v[j]; q += v[j] * v[j]; }
  for (int o = 16; o; o >>= 1) {
    s += __shfl_xor(s, o, 32);
    q += __shfl_xor(q, o, 32);
  }
  const float mu  = s * (1.f / CC);
  const float var = q * (1.f / CC) - mu * mu;
  const float r   = rsqrtf(var + EPS_LN);
#pragma unroll
  for (int j = 0; j < 8; j++) {
    const int c = lane * 8 + j;
    const float y = (v[j] - mu) * r * g[c] + b[c];
    if (MODE == 0)
      outb[row * (2 * CC) + CC + c] = (bf16)y;
    else
      outf[row * CC + c] = resid[row * CC + c] + y;
  }
}

// ---------------------------------------------------------------------------
extern "C" void kernel_launch(void* const* d_in, const int* in_sizes, int n_in,
                              void* d_out, int out_size, void* d_ws, size_t ws_size,
                              hipStream_t stream) {
  const float* x   = (const float*)d_in[0];
  const float* src = (const float*)d_in[1];
  const float* Wq  = (const float*)d_in[2];
  const float* Wk  = (const float*)d_in[3];
  const float* Wv  = (const float*)d_in[4];
  const float* Wm  = (const float*)d_in[5];
  const float* W1  = (const float*)d_in[6];
  const float* W2  = (const float*)d_in[7];
  const float* g1  = (const float*)d_in[8];
  const float* b1  = (const float*)d_in[9];
  const float* g2  = (const float*)d_in[10];
  const float* b2  = (const float*)d_in[11];
  float* out = (float*)d_out;

  char* ws = (char*)d_ws;
  size_t off = 0;
  auto alloc = [&](size_t bytes) -> void* {
    void* p = ws + off;
    off = (off + bytes + 255) & ~(size_t)255;
    return p;
  };

  bf16* wq_t  = (bf16*)alloc((size_t)CC * CC * 2);
  bf16* wk_t  = (bf16*)alloc((size_t)CC * CC * 2);
  bf16* wv_t  = (bf16*)alloc((size_t)CC * CC * 2);
  bf16* wm_t  = (bf16*)alloc((size_t)CC * CC * 2);
  bf16* w1_t  = (bf16*)alloc((size_t)2 * CC * 2 * CC * 2); // [512][512]
  bf16* w2_t  = (bf16*)alloc((size_t)CC * 2 * CC * 2);     // [256][512]
  bf16* sb    = (bf16*)alloc((size_t)MTOK * CC * 2);
  bf16* hcat  = (bf16*)alloc((size_t)MTOK * 2 * CC * 2);   // [x | LN(msg)]
  bf16* Qb    = (bf16*)alloc((size_t)MTOK * CC * 2);
  bf16* Kb    = (bf16*)alloc((size_t)MTOK * CC * 2);
  bf16* Vnb   = (bf16*)alloc((size_t)MTOK * CC * 2);
  float* KV   = (float*)alloc((size_t)NB * HH * DD * DD * 4);
  float* Ksum = (float*)alloc((size_t)NB * HH * DD * 4);
  bf16* msgA  = (bf16*)alloc((size_t)MTOK * CC * 2);
  float* mf32 = (float*)alloc((size_t)MTOK * CC * 4);      // msgB, reused as m2
  bf16* h1    = (bf16*)alloc((size_t)MTOK * 2 * CC * 2);

  // Phase A: weight transpose-convert + activation convert
  k_convert_wt<<<(CC * CC + 255) / 256, 256, 0, stream>>>(Wq, wq_t, CC, CC);
  k_convert_wt<<<(CC * CC + 255) / 256, 256, 0, stream>>>(Wk, wk_t, CC, CC);
  k_convert_wt<<<(CC * CC + 255) / 256, 256, 0, stream>>>(Wv, wv_t, CC, CC);
  k_convert_wt<<<(CC * CC + 255) / 256, 256, 0, stream>>>(Wm, wm_t, CC, CC);
  k_convert_wt<<<(4 * CC * CC + 255) / 256, 256, 0, stream>>>(W1, w1_t, 2 * CC, 2 * CC);
  k_convert_wt<<<(2 * CC * CC + 255) / 256, 256, 0, stream>>>(W2, w2_t, 2 * CC, CC);
  k_convert_x_to_hcat<<<(MTOK * CC) / 256, 256, 0, stream>>>(x, hcat);
  k_convert<<<(MTOK * CC) / 256, 256, 0, stream>>>(src, sb, (long)MTOK * CC);
  hipMemsetAsync(KV, 0, (size_t)NB * HH * DD * DD * 4, stream);
  hipMemsetAsync(Ksum, 0, (size_t)NB * HH * DD * 4, stream);

  // Phase B: projections (WMMA bf16) with fused epilogues
  dim3 g256(MTOK / 128, CC / 64);
  k_gemm<1><<<g256, 256, 0, stream>>>(hcat, 2 * CC, wq_t, CC, Qb, CC, CC, 0.f);   // Q=elu(xWq)+1
  k_gemm<1><<<g256, 256, 0, stream>>>(sb, CC, wk_t, CC, Kb, CC, CC, 0.f);         // K=elu(sWk)+1
  k_gemm<3><<<g256, 256, 0, stream>>>(sb, CC, wv_t, CC, Vnb, CC, CC, 1.f / SSEQ); // Vn=sWv/S

  // Phase C: linear attention core
  k_kv<<<dim3(32, 32), 256, 0, stream>>>(Kb, Vnb, KV, Ksum);
  k_msg<<<NB * (LL / 32), 256, 0, stream>>>(Qb, KV, Ksum, msgA);

  // Phase D: merge + LN1 into concat buffer
  k_gemm<0><<<g256, 256, 0, stream>>>(msgA, CC, wm_t, CC, mf32, CC, CC, 0.f);
  k_ln<0><<<MTOK / 8, 256, 0, stream>>>(mf32, g1, b1, hcat, nullptr, nullptr);

  // Phase E: MLP
  dim3 g512(MTOK / 128, (2 * CC) / 64);
  k_gemm<2><<<g512, 256, 0, stream>>>(hcat, 2 * CC, w1_t, 2 * CC, h1, 2 * CC, 2 * CC, 0.f);
  k_gemm<0><<<g256, 256, 0, stream>>>(h1, 2 * CC, w2_t, 2 * CC, mf32, CC, 2 * CC, 0.f);

  // Phase F: LN2 + residual
  k_ln<1><<<MTOK / 8, 256, 0, stream>>>(mf32, g2, b2, nullptr, x, out);
}
